// MD17EGNNAdapter_13950053777625
// MI455X (gfx1250) — matrix operand order repaired and checked by reading, not compiled, not run on hardware
//
#include <hip/hip_runtime.h>
#include <hip/hip_bf16.h>
#include <stdint.h>

typedef __attribute__((ext_vector_type(16))) _Float16 v16h;
typedef __attribute__((ext_vector_type(8)))  float    v8f;

union FragU { uint32_t u[8]; v16h h; };

#define BB 64
#define NN 64
#define DD 128
#define BN (BB*NN)

// fast silu: x * v_rcp_f32(1 + exp(-x)) — avoids the IEEE v_div_scale/fixup chain
__device__ __forceinline__ float fast_silu(float x) {
    return x * __builtin_amdgcn_rcpf(1.f + __expf(-x));
}

// ---------------- weight conversion: ew2 (L,D,D) f32 -> ew2T (L,D,D) f16 transposed ----------------
__global__ void cvt_kernel(const float* __restrict__ ew2, _Float16* __restrict__ ew2T) {
    int idx = blockIdx.x * blockDim.x + threadIdx.x;
    if (idx >= 4 * DD * DD) return;
    int l = idx >> 14;
    int r = idx & 16383;
    int dout = r >> 7, din = r & 127;
    // ew2T[l][dout][din] = ew2[l][din][dout]
    ew2T[idx] = (_Float16)ew2[l * DD * DD + din * DD + dout];
}

// ---------------- h = embed[atomic_numbers] ----------------
__global__ __launch_bounds__(128) void embed_kernel(const int* __restrict__ an,
                                                    const float* __restrict__ emb,
                                                    float* __restrict__ h) {
    int row = blockIdx.x;
    int d = threadIdx.x;
    h[row * DD + d] = emb[an[row] * DD + d];
}

// ---------------- dist_sq[b,i,j] ----------------
__global__ void dist_kernel(const float* __restrict__ pos, float* __restrict__ dist) {
    int idx = blockIdx.x * blockDim.x + threadIdx.x;
    if (idx >= BB * NN * NN) return;
    int j = idx & 63, i = (idx >> 6) & 63, b = idx >> 12;
    const float* pi = pos + (b * NN + i) * 3;
    const float* pj = pos + (b * NN + j) * 3;
    float dx = pi[0] - pj[0], dy = pi[1] - pj[1], dz = pi[2] - pj[2];
    dist[idx] = dx * dx + dy * dy + dz * dz;
}

// ---------------- hwa = h@wa, hwb = h@wb ----------------
__global__ __launch_bounds__(128) void proj_kernel(const float* __restrict__ h,
                                                   const float* __restrict__ wa,
                                                   const float* __restrict__ wb,
                                                   float* __restrict__ hwa,
                                                   float* __restrict__ hwb) {
    __shared__ float hrow[DD];
    int row = blockIdx.x, d = threadIdx.x;
    hrow[d] = h[row * DD + d];
    __syncthreads();
    float aacc = 0.f, bacc = 0.f;
    for (int k = 0; k < DD; ++k) {
        float hv = hrow[k];
        aacc += hv * wa[k * DD + d];
        bacc += hv * wb[k * DD + d];
    }
    hwa[row * DD + d] = aacc;
    hwb[row * DD + d] = bacc;
}

// ---------------- fused edge MLP + mean-over-j, one block per (b,i) ----------------
__global__ __launch_bounds__(256) void edge_kernel(const float* __restrict__ hwa,
                                                   const float* __restrict__ hwb,
                                                   const float* __restrict__ dist,
                                                   const float* __restrict__ wd,
                                                   const float* __restrict__ eb1,
                                                   const _Float16* __restrict__ ew2T,
                                                   const float* __restrict__ eb2,
                                                   float* __restrict__ m_i) {
    __shared__ _Float16 aT[NN * DD];      // silu(pre) tile, row-major [j][d], fp16 (16 KB)
    __shared__ _Float16 bT[DD * DD];      // ew2 transposed [n][k], fp16 (32 KB)
    __shared__ float msum[DD];
    __shared__ float rowA[DD];
    __shared__ float wdv[DD];
    __shared__ float eb1v[DD];

    const int tid = threadIdx.x;
    const int bi = blockIdx.x;           // b*64 + i
    const int b = bi >> 6;
    const int i = bi & 63;

    // hint: pull weight tile toward this WGP early (global_prefetch_b8)
    __builtin_prefetch(ew2T + tid * 64, 0, 1);

    if (tid < DD) {
        rowA[tid] = hwa[bi * DD + tid];
        wdv[tid]  = wd[tid];
        eb1v[tid] = eb1[tid];
        msum[tid] = 0.f;
    }
    // stage ew2T into LDS (16B vector copies)
    {
        const uint4* src = (const uint4*)ew2T;
        uint4* dst = (uint4*)bT;
        for (int e = tid; e < (DD * DD * 2) / 16; e += 256) dst[e] = src[e];
    }
    __syncthreads();

    // build A = silu(pre) in fp16, pre[j][d] = hwa[i,d] + hwb[j,d] + dist*wd[d] + eb1[d]
    for (int e = tid; e < NN * (DD / 2); e += 256) {
        int j = e >> 6;
        int d0 = (e & 63) * 2;
        float ds = dist[(b * NN + i) * NN + j];
        const float* hb = hwb + (b * NN + j) * DD;
        float x0 = rowA[d0]     + hb[d0]     + ds * wdv[d0]     + eb1v[d0];
        float x1 = rowA[d0 + 1] + hb[d0 + 1] + ds * wdv[d0 + 1] + eb1v[d0 + 1];
        aT[j * DD + d0]     = (_Float16)fast_silu(x0);
        aT[j * DD + d0 + 1] = (_Float16)fast_silu(x1);
    }
    __syncthreads();

    // WMMA: (64 x 128) @ (128 x 128), 8 waves, each wave 4 C-tiles of 16x16
    const int wave = tid >> 5;
    const int lane = tid & 31;
    const int mt = wave & 3;             // M tile (j rows)
    const int hi = lane >> 4;
    const int ln = lane & 15;
    const int arow = mt * 16 + ln;       // A row per ISA layout: lanes 0-15/16-31 -> M = lane&15

    // A row is fixed per wave: hoist all 4 K-chunk A fragments (loaded once)
    FragU afr[4];
#pragma unroll
    for (int kt = 0; kt < 4; ++kt) {
#pragma unroll
        for (int v = 0; v < 8; ++v) {
            // A (16-bit 16x32): word v holds K pair (v&3)*2 + (v>>2)*16 + hi*8
            int ka = ((v & 3) * 2) + ((v >> 2) * 16) + hi * 8 + kt * 32;
            afr[kt].u[v] = *(const uint32_t*)&aT[arow * DD + ka];
        }
    }

#pragma unroll
    for (int tt = 0; tt < 4; ++tt) {
        const int nt = (wave >> 2) + tt * 2;   // N tile
        const int ncol = nt * 16 + ln;
        v8f acc = {};
#pragma unroll
        for (int kt = 0; kt < 4; ++kt) {       // K = 128 in chunks of 32
            FragU bfr;
#pragma unroll
            for (int v = 0; v < 8; ++v) {
                // B (16-bit 32x16): lane n = lane&15, word v holds K pair (v&3)*2 + (v>>2)*16 + hi*8
                int kb = ((v & 3) * 2) + ((v >> 2) * 16) + hi * 8 + kt * 32;
                bfr.u[v] = *(const uint32_t*)&bT[ncol * DD + kb];
            }
            acc = __builtin_amdgcn_wmma_f32_16x16x32_f16(
                false, afr[kt].h, false, bfr.h, (short)0, acc, false, false);
        }
        // epilogue: + eb2, silu, column-sum (mean over j)
        float bias = eb2[ncol];
        float csum = 0.f;
#pragma unroll
        for (int r = 0; r < 8; ++r) {
            csum += fast_silu(acc[r] + bias);
        }
        csum += __shfl_xor(csum, 16, 32);      // combine M halves (rows r and r+8)
        if (lane < 16) atomicAdd(&msum[ncol], csum);   // combine across 4 M tiles
    }
    __syncthreads();
    if (tid < DD) m_i[bi * DD + tid] = msum[tid] * (1.0f / (float)NN);
}

// ---------------- t = silu(h@nw1a + m_i@nw1b + nb1) ----------------
__global__ __launch_bounds__(128) void node1_kernel(const float* __restrict__ h,
                                                    const float* __restrict__ mi,
                                                    const float* __restrict__ nw1,
                                                    const float* __restrict__ nb1,
                                                    float* __restrict__ t) {
    __shared__ float hrow[DD];
    __shared__ float mrow[DD];
    int row = blockIdx.x, d = threadIdx.x;
    hrow[d] = h[row * DD + d];
    mrow[d] = mi[row * DD + d];
    __syncthreads();
    float acc = nb1[d];
    for (int k = 0; k < DD; ++k) {
        acc += hrow[k] * nw1[k * DD + d] + mrow[k] * nw1[(DD + k) * DD + d];
    }
    t[row * DD + d] = fast_silu(acc);
}

// ---------------- h = LN(h + t@nw2 + nb2) * g + b ----------------
__global__ __launch_bounds__(128) void node2_kernel(const float* __restrict__ t,
                                                    const float* __restrict__ h_in,
                                                    const float* __restrict__ nw2,
                                                    const float* __restrict__ nb2,
                                                    const float* __restrict__ g,
                                                    const float* __restrict__ be,
                                                    float* __restrict__ h_out) {
    __shared__ float trow[DD];
    __shared__ float red[DD];
    int row = blockIdx.x, d = threadIdx.x;
    trow[d] = t[row * DD + d];
    __syncthreads();
    float acc = nb2[d];
    for (int k = 0; k < DD; ++k) acc += trow[k] * nw2[k * DD + d];
    float x = h_in[row * DD + d] + acc;

    red[d] = x; __syncthreads();
    for (int s = 64; s > 0; s >>= 1) { if (d < s) red[d] += red[d + s]; __syncthreads(); }
    float mu = red[0] * (1.f / (float)DD);
    __syncthreads();
    float c = x - mu;
    red[d] = c * c; __syncthreads();
    for (int s = 64; s > 0; s >>= 1) { if (d < s) red[d] += red[d + s]; __syncthreads(); }
    float var = red[0] * (1.f / (float)DD);
    h_out[row * DD + d] = c * rsqrtf(var + 1e-5f) * g[d] + be[d];
}

// ---------------- out = h @ out_w + out_b ----------------
__global__ void out_kernel(const float* __restrict__ h, const float* __restrict__ ow,
                           const float* __restrict__ ob, float* __restrict__ out) {
    int idx = blockIdx.x * blockDim.x + threadIdx.x;
    if (idx >= BN * 3) return;
    int row = idx / 3, c = idx % 3;
    float acc = ob[c];
    for (int k = 0; k < DD; ++k) acc += h[row * DD + k] * ow[k * 3 + c];
    out[idx] = acc;
}

extern "C" void kernel_launch(void* const* d_in, const int* in_sizes, int n_in,
                              void* d_out, int out_size, void* d_ws, size_t ws_size,
                              hipStream_t stream) {
    (void)in_sizes; (void)n_in; (void)out_size; (void)ws_size;
    const float* positions = (const float*)d_in[0];
    const int*   an        = (const int*)d_in[1];
    const float* emb       = (const float*)d_in[2];
    const float* ew1       = (const float*)d_in[3];
    const float* eb1       = (const float*)d_in[4];
    const float* ew2       = (const float*)d_in[5];
    const float* eb2       = (const float*)d_in[6];
    const float* nw1       = (const float*)d_in[7];
    const float* nb1       = (const float*)d_in[8];
    const float* nw2       = (const float*)d_in[9];
    const float* nb2       = (const float*)d_in[10];
    const float* ln_g      = (const float*)d_in[11];
    const float* ln_b      = (const float*)d_in[12];
    const float* out_w     = (const float*)d_in[13];
    const float* out_b     = (const float*)d_in[14];
    float* out = (float*)d_out;

    float* ws = (float*)d_ws;
    float* h    = ws;  ws += BN * DD;
    float* hwa  = ws;  ws += BN * DD;
    float* hwb  = ws;  ws += BN * DD;
    float* mi   = ws;  ws += BN * DD;
    float* tb   = ws;  ws += BN * DD;
    float* dist = ws;  ws += BB * NN * NN;
    _Float16* ew2T = (_Float16*)ws;      // L*D*D halfs

    cvt_kernel<<<(4 * DD * DD + 255) / 256, 256, 0, stream>>>(ew2, ew2T);
    embed_kernel<<<BN, 128, 0, stream>>>(an, emb, h);
    dist_kernel<<<(BB * NN * NN + 255) / 256, 256, 0, stream>>>(positions, dist);

    for (int l = 0; l < 4; ++l) {
        const float* ew1l = ew1 + l * (2 * DD + 1) * DD;
        const float* wa = ew1l;
        const float* wb = ew1l + DD * DD;
        const float* wd = ew1l + 2 * DD * DD;
        proj_kernel<<<BN, 128, 0, stream>>>(h, wa, wb, hwa, hwb);
        edge_kernel<<<BN, 256, 0, stream>>>(hwa, hwb, dist, wd, eb1 + l * DD,
                                            ew2T + l * DD * DD, eb2 + l * DD, mi);
        node1_kernel<<<BN, 128, 0, stream>>>(h, mi, nw1 + l * 2 * DD * DD, nb1 + l * DD, tb);
        node2_kernel<<<BN, 128, 0, stream>>>(tb, h, nw2 + l * DD * DD, nb2 + l * DD,
                                             ln_g + l * DD, ln_b + l * DD, h);
    }
    out_kernel<<<(BN * 3 + 255) / 256, 256, 0, stream>>>(h, out_w, out_b, out);
}
